// GATLayer_39479339385231
// MI455X (gfx1250) — matrix operand order repaired and checked by reading, not compiled, not run on hardware
//
#include <hip/hip_runtime.h>
#include <hip/hip_bf16.h>

// ---------------------------------------------------------------------------
// GAT layer for MI455X (gfx1250, wave32, WMMA).
//
// Pipeline (all on `stream`):
//   K0 init      : zero H_out, asum; amax_bits = INT_MIN
//   K1 fold_we   : WattE[k][h] = sum_d W_e[k, h*16+d] * att_edge[h,d]   (32x4)
//   K2 ht_wmma   : Ht = H @ W   via v_wmma_f32_16x16x32_f16 (f32 accum)
//   K3 node_sc   : s_src[n,h], s_dst[n,h] = <Ht[n,h,:], att_{src,dst}[h]>
//   K4 edge_log  : alpha = lrelu(s_src[src]+s_dst[dst]+E@WattE); atomicMax amax
//   K5 edge_exp  : alpha = exp(alpha - amax[dst]); atomicAdd asum[dst]
//   K6 norm      : alpha /= (asum[dst] + 1e-16)
//   K7 scatter   : H_out[dst] += Ht[src] * alpha
// ---------------------------------------------------------------------------

typedef __attribute__((ext_vector_type(16))) _Float16 v16h;
typedef __attribute__((ext_vector_type(8)))  float    v8f;

#define DIN   128
#define DH     64   // HEADS * D_OUT
#define NHEAD   4
#define DOUT   16
#define WSTR  136   // padded LDS row stride (halfs) for transposed W

// ---- order-preserving float <-> int mapping for atomicMax on floats --------
__device__ __forceinline__ int f2ord(float f) {
  int b = __float_as_int(f);
  return b >= 0 ? b : (b ^ 0x7FFFFFFF);
}
__device__ __forceinline__ float ord2f(int i) {
  return __int_as_float(i >= 0 ? i : (i ^ 0x7FFFFFFF));
}

// ---------------------------------------------------------------------------
__global__ void __launch_bounds__(256)
k_init(float* __restrict__ out, int n_out,
       int* __restrict__ amax, float* __restrict__ asum, int n_nh) {
  int i = blockIdx.x * blockDim.x + threadIdx.x;
  int stride = gridDim.x * blockDim.x;
  for (int idx = i; idx < n_out; idx += stride) out[idx] = 0.0f;
  for (int idx = i; idx < n_nh; idx += stride) {
    amax[idx] = 0x80000000;   // below f2ord of any finite float
    asum[idx] = 0.0f;
  }
}

// ---------------------------------------------------------------------------
__global__ void __launch_bounds__(128)
k_fold_we(const float* __restrict__ W_e, const float* __restrict__ att_edge,
          float* __restrict__ WattE) {
  int idx = threadIdx.x;              // 0..127  -> (k, h)
  if (idx >= 32 * NHEAD) return;
  int k = idx >> 2, h = idx & 3;
  float s = 0.0f;
  #pragma unroll
  for (int d = 0; d < DOUT; ++d)
    s += W_e[k * DH + h * DOUT + d] * att_edge[h * DOUT + d];
  WattE[idx] = s;                     // layout [k][h]
}

// ---------------------------------------------------------------------------
// Ht = H @ W,  tiles of 16 rows per wave, 8 waves per block.
__global__ void __launch_bounds__(256)
k_ht_wmma(const float* __restrict__ H, const float* __restrict__ W,
          float* __restrict__ Ht, int n_nodes) {
  __shared__ _Float16 Wlds[DH * WSTR];   // W transposed: [col][k], padded

  // cooperative load + convert + transpose of W (128x64 f32 -> 64x128 f16)
  for (int idx = threadIdx.x; idx < DIN * DH; idx += blockDim.x) {
    int k = idx >> 6, c = idx & 63;      // W is [k][c]
    Wlds[c * WSTR + k] = (_Float16)W[idx];
  }
  __syncthreads();

  int wave = threadIdx.x >> 5;           // 0..7
  int lane = threadIdx.x & 31;
  int n_tiles = (n_nodes + 15) >> 4;
  int tile = blockIdx.x * 8 + wave;
  if (tile >= n_tiles) return;           // wave-uniform exit (after barrier)

  int m0      = tile << 4;
  int halfsel = lane >> 4;               // 0: lanes 0-15, 1: lanes 16-31
  int lrow    = lane & 15;

  int arow = m0 + lrow;                  // A-fragment source row
  if (arow >= n_nodes) arow = n_nodes - 1;   // clamp (EXEC must stay all-1)
  const float* hrow = H + (size_t)arow * DIN;

  v8f acc0 = {}, acc1 = {}, acc2 = {}, acc3 = {};

  for (int k0 = 0; k0 < DIN; k0 += 32) {
    // ---- A fragment: 16x32 f16, k = 16*halfsel + s (same order as B) -----------------------------
    v16h a;
    int kbase = k0 + halfsel * 16;
    #pragma unroll
    for (int i = 0; i < 8; ++i) {
      a[i]     = (_Float16)hrow[kbase + i];
      a[i + 8] = (_Float16)hrow[kbase + 8 + i];
    }
    // ---- B fragments (4 N-tiles): 32x16 f16, column per lane --------------
    int kb = k0 + halfsel * 16;
#define DO_TILE(T, ACC)                                                      \
    {                                                                        \
      v16h b;                                                                \
      const _Float16* wp = &Wlds[(T * 16 + lrow) * WSTR + kb];               \
      _Pragma("unroll")                                                      \
      for (int i = 0; i < 16; ++i) b[i] = wp[i];                             \
      ACC = __builtin_amdgcn_wmma_f32_16x16x32_f16(                          \
          false, a, false, b, (short)0, ACC, false, false);                  \
    }
    DO_TILE(0, acc0)
    DO_TILE(1, acc1)
    DO_TILE(2, acc2)
    DO_TILE(3, acc3)
#undef DO_TILE
  }

  // ---- store D: VGPR r -> row m0 + halfsel*8 + r, col lrow + t*16 ---------
  int row0 = m0 + halfsel * 8;
  if (m0 + 16 <= n_nodes) {
    // full tile (the common case): unguarded, branch-free stores
    #pragma unroll
    for (int r = 0; r < 8; ++r) {
      float* o = Ht + (size_t)(row0 + r) * DH + lrow;
      o[0]  = acc0[r];
      o[16] = acc1[r];
      o[32] = acc2[r];
      o[48] = acc3[r];
    }
  } else {
    #pragma unroll
    for (int r = 0; r < 8; ++r) {
      int row = row0 + r;
      if (row < n_nodes) {
        float* o = Ht + (size_t)row * DH + lrow;
        o[0]  = acc0[r];
        o[16] = acc1[r];
        o[32] = acc2[r];
        o[48] = acc3[r];
      }
    }
  }
}

// ---------------------------------------------------------------------------
__global__ void __launch_bounds__(256)
k_node_scores(const float* __restrict__ Ht,
              const float* __restrict__ att_src, const float* __restrict__ att_dst,
              float* __restrict__ s_src, float* __restrict__ s_dst, int n_nodes) {
  int idx = blockIdx.x * blockDim.x + threadIdx.x;   // (node, h)
  if (idx >= n_nodes * NHEAD) return;
  int node = idx >> 2, h = idx & 3;
  const float* p = Ht + (size_t)node * DH + h * DOUT;
  float ss = 0.0f, sd = 0.0f;
  #pragma unroll
  for (int d = 0; d < DOUT; ++d) {
    float v = p[d];
    ss += v * att_src[h * DOUT + d];
    sd += v * att_dst[h * DOUT + d];
  }
  s_src[idx] = ss;
  s_dst[idx] = sd;
}

// ---------------------------------------------------------------------------
__global__ void __launch_bounds__(256)
k_edge_logits(const int* __restrict__ src, const int* __restrict__ dst,
              const float* __restrict__ E, const float* __restrict__ s_src,
              const float* __restrict__ s_dst, const float* __restrict__ WattE,
              float* __restrict__ alpha, int* __restrict__ amax, int n_edges) {
  __shared__ float wl[32 * NHEAD];
  if (threadIdx.x < 32 * NHEAD) wl[threadIdx.x] = WattE[threadIdx.x];
  __syncthreads();

  int e = blockIdx.x * blockDim.x + threadIdx.x;
  if (e >= n_edges) return;
  int s = src[e], d = dst[e];

  const float4* ep = (const float4*)(E + (size_t)e * 32);
  float a0 = 0.f, a1 = 0.f, a2 = 0.f, a3 = 0.f;
  #pragma unroll
  for (int q = 0; q < 8; ++q) {           // 32 k-values, float4 at a time
    float4 v = ep[q];
    const float* w = &wl[q * 16];
    a0 += v.x * w[0] + v.y * w[4] + v.z * w[8]  + v.w * w[12];
    a1 += v.x * w[1] + v.y * w[5] + v.z * w[9]  + v.w * w[13];
    a2 += v.x * w[2] + v.y * w[6] + v.z * w[10] + v.w * w[14];
    a3 += v.x * w[3] + v.y * w[7] + v.z * w[11] + v.w * w[15];
  }
  float es[4] = {a0, a1, a2, a3};
  #pragma unroll
  for (int h = 0; h < NHEAD; ++h) {
    float a = s_src[s * NHEAD + h] + s_dst[d * NHEAD + h] + es[h];
    a = a > 0.0f ? a : 0.2f * a;          // leaky relu
    alpha[(size_t)e * NHEAD + h] = a;
    atomicMax(&amax[d * NHEAD + h], f2ord(a));
  }
}

// ---------------------------------------------------------------------------
__global__ void __launch_bounds__(256)
k_edge_exp(const int* __restrict__ dst, float* __restrict__ alpha,
           const int* __restrict__ amax, float* __restrict__ asum, int n_edges) {
  int idx = blockIdx.x * blockDim.x + threadIdx.x;   // (e, h)
  if (idx >= n_edges * NHEAD) return;
  int e = idx >> 2, h = idx & 3;
  int d = dst[e];
  float m = ord2f(amax[d * NHEAD + h]);
  float v = __expf(alpha[idx] - m);
  alpha[idx] = v;
  atomicAdd(&asum[d * NHEAD + h], v);
}

// ---------------------------------------------------------------------------
__global__ void __launch_bounds__(256)
k_normalize(const int* __restrict__ dst, float* __restrict__ alpha,
            const float* __restrict__ asum, int n_eh) {
  int idx = blockIdx.x * blockDim.x + threadIdx.x;   // (e, h)
  if (idx >= n_eh) return;
  int e = idx >> 2, h = idx & 3;
  int d = dst[e];
  alpha[idx] = alpha[idx] / (asum[d * NHEAD + h] + 1e-16f);
}

// ---------------------------------------------------------------------------
__global__ void __launch_bounds__(256)
k_scatter(const int* __restrict__ src, const int* __restrict__ dst,
          const float* __restrict__ Ht, const float* __restrict__ alpha,
          float* __restrict__ out, long long total) {
  long long idx = (long long)blockIdx.x * blockDim.x + threadIdx.x; // (e, j)
  if (idx >= total) return;
  int e = (int)(idx >> 6);
  int j = (int)(idx & 63);
  int h = j >> 4;
  int s = src[e], d = dst[e];
  float w = alpha[(size_t)e * NHEAD + h];
  atomicAdd(&out[(size_t)d * DH + j], Ht[(size_t)s * DH + j] * w);
}

// ---------------------------------------------------------------------------
extern "C" void kernel_launch(void* const* d_in, const int* in_sizes, int n_in,
                              void* d_out, int out_size, void* d_ws, size_t ws_size,
                              hipStream_t stream) {
  const float* H        = (const float*)d_in[0];
  const int*   ei       = (const int*)  d_in[1];
  const float* E        = (const float*)d_in[2];
  const float* W        = (const float*)d_in[3];
  const float* W_e      = (const float*)d_in[4];
  const float* att_src  = (const float*)d_in[5];
  const float* att_dst  = (const float*)d_in[6];
  const float* att_edge = (const float*)d_in[7];

  const int n_nodes = in_sizes[0] / DIN;
  const int n_edges = in_sizes[1] / 2;
  const int* src = ei;
  const int* dst = ei + n_edges;

  float* out = (float*)d_out;

  // ---- workspace carve-up (float/int units, 256-elt aligned) --------------
  float* ws = (float*)d_ws;
  size_t off = 0;
  auto take = [&](size_t n) { float* p = ws + off; off += (n + 255) & ~(size_t)255; return p; };
  float* Ht    = take((size_t)n_nodes * DH);
  float* s_src = take((size_t)n_nodes * NHEAD);
  float* s_dst = take((size_t)n_nodes * NHEAD);
  float* WattE = take(32 * NHEAD);
  int*   amax  = (int*)take((size_t)n_nodes * NHEAD);
  float* asum  = take((size_t)n_nodes * NHEAD);
  float* alpha = take((size_t)n_edges * NHEAD);
  (void)ws_size;

  const int n_nh = n_nodes * NHEAD;
  const int n_eh = n_edges * NHEAD;

  // K0: init out/amax/asum
  {
    int work = out_size > n_nh ? out_size : n_nh;
    int blocks = (work + 255) / 256;
    if (blocks > 16384) blocks = 16384;
    k_init<<<blocks, 256, 0, stream>>>(out, out_size, amax, asum, n_nh);
  }
  // K1: fold W_e with att_edge
  k_fold_we<<<1, 128, 0, stream>>>(W_e, att_edge, WattE);
  // K2: Ht = H @ W  (WMMA)
  {
    int n_tiles = (n_nodes + 15) / 16;
    int blocks  = (n_tiles + 7) / 8;
    k_ht_wmma<<<blocks, 256, 0, stream>>>(H, W, Ht, n_nodes);
  }
  // K3: per-node attention scores
  k_node_scores<<<(n_nh + 255) / 256, 256, 0, stream>>>(Ht, att_src, att_dst,
                                                        s_src, s_dst, n_nodes);
  // K4: edge logits + segment max
  k_edge_logits<<<(n_edges + 255) / 256, 256, 0, stream>>>(src, dst, E, s_src,
                                                           s_dst, WattE, alpha,
                                                           amax, n_edges);
  // K5: exp + segment sum
  k_edge_exp<<<(n_eh + 255) / 256, 256, 0, stream>>>(dst, alpha, amax, asum,
                                                     n_edges);
  // K6: normalize alpha once per (edge, head)
  k_normalize<<<(n_eh + 255) / 256, 256, 0, stream>>>(dst, alpha, asum, n_eh);
  // K7: scatter-add messages
  {
    long long total = (long long)n_edges * DH;
    long long blocks = (total + 255) / 256;
    k_scatter<<<(int)blocks, 256, 0, stream>>>(src, dst, Ht, alpha, out, total);
  }
}
